// GAT_54202487276064
// MI455X (gfx1250) — matrix operand order repaired and checked by reading, not compiled, not run on hardware
//
#include <hip/hip_runtime.h>
#include <hip/hip_bf16.h>

// ---------------------------------------------------------------------------
// GAT pipeline for MI455X (gfx1250, wave32, WMMA).
// Big GEMMs -> v_wmma_f32_16x16x32_bf16 (bf16 in, f32 accumulate), staged via
// async global->LDS (ASYNCcnt) with double buffering when available.
// Elementwise / gather / segment-sum -> memory-bound kernels w/ HW f32 atomics.
// ---------------------------------------------------------------------------

#define NN 10000          // nodes
#define NE 80000          // edges
#define DD 512            // feature dim
#define K2 1024           // 2*D (GEMM K)
#define LL 5              // layers

typedef __attribute__((ext_vector_type(16))) __bf16 v16bf;
typedef __attribute__((ext_vector_type(8)))  float  v8f;
typedef int v4i_t __attribute__((vector_size(16)));

union FragU {
    uint4 q[2];
    v16bf v;
};

#if __has_builtin(__builtin_amdgcn_global_load_async_to_lds_b128)
#define HAVE_ASYNC 1
#else
#define HAVE_ASYNC 0
#endif

#if HAVE_ASYNC
#if __has_builtin(__builtin_amdgcn_s_wait_asynccnt)
#define WAIT_ASYNC(n) __builtin_amdgcn_s_wait_asynccnt(n)
#else
#define WAIT_ASYNC(n) asm volatile("s_wait_asynccnt %0" ::"n"(n) : "memory")
#endif
// builtin expects (int4 addrspace(1)*, int4 addrspace(3)*, imm offset, imm cpol)
#define AS1P(p) ((__attribute__((address_space(1))) v4i_t*)(uintptr_t)(p))
#define AS3P(p) ((__attribute__((address_space(3))) v4i_t*)(unsigned)(uintptr_t)(p))
#endif

__device__ __forceinline__ unsigned short f2bf(float x) {
    unsigned int u = __float_as_uint(x);
    u += 0x7FFFu + ((u >> 16) & 1u);      // round to nearest even
    return (unsigned short)(u >> 16);
}

__device__ __forceinline__ void atomicMaxF(float* a, float v) {
    // sign-split trick: avoids CAS loop, uses native integer atomics.
    if (v >= 0.0f) atomicMax((int*)a, __float_as_int(v));
    else           atomicMin((unsigned int*)a, __float_as_uint(v));
}

__device__ __forceinline__ float leaky2(float x) {
    // leaky(leaky(x, 0.1), 0.1): negative branch -> 0.01*x
    return x >= 0.0f ? x : 0.01f * x;
}

// ---------------------------------------------------------------------------
// fill
// ---------------------------------------------------------------------------
__global__ void fill_k(float* __restrict__ p, float v, int n) {
    int i = blockIdx.x * 256 + threadIdx.x;
    if (i < n) p[i] = v;
}

// ---------------------------------------------------------------------------
// W[k,n] f32 -> Wt[n,k] bf16  (so B fragments are contiguous in K)
// ---------------------------------------------------------------------------
__global__ void tcast_k(const float* __restrict__ W, unsigned short* __restrict__ Wt,
                        int K, int N) {
    int i = blockIdx.x * 256 + threadIdx.x;
    if (i >= K * N) return;
    int k = i / N, n = i % N;
    Wt[(size_t)n * K + k] = f2bf(W[i]);
}

// ---------------------------------------------------------------------------
// cat[e, 0:512]   = bf16(ex[e,:])
// cat[e, 512:1024]= bf16(nx[idx[e],:])
// ---------------------------------------------------------------------------
__global__ void cat_edge_k(const float* __restrict__ ex, const float* __restrict__ nx,
                           const int* __restrict__ idx, unsigned short* __restrict__ cat) {
    int i = blockIdx.x * 256 + threadIdx.x;           // over NE*DD
    if (i >= NE * DD) return;
    int e = i / DD, d = i - e * DD;
    size_t o = (size_t)e * K2 + d;
    cat[o]      = f2bf(ex[i]);
    cat[o + DD] = f2bf(nx[(size_t)idx[e] * DD + d]);
}

// ---------------------------------------------------------------------------
// column max over M rows (partial per block, then sign-split atomic)
// ---------------------------------------------------------------------------
__global__ void colmax_k(const float* __restrict__ A, float* __restrict__ mx, int M) {
    int r0 = blockIdx.x * 256;
    int c  = threadIdx.x * 2;                         // 256 threads * 2 cols = 512
    float m0 = -__builtin_huge_valf(), m1 = m0;
    for (int i = 0; i < 256; ++i) {
        int r = r0 + i;
        if (r < M) {
            float2 v = *(const float2*)(A + (size_t)r * DD + c);
            m0 = fmaxf(m0, v.x);
            m1 = fmaxf(m1, v.y);
        }
    }
    atomicMaxF(&mx[c], m0);
    atomicMaxF(&mx[c + 1], m1);
}

// ---------------------------------------------------------------------------
// ap = exp(ap - colmax)  (in place), z[dst] += ap   (segment sum)
// ---------------------------------------------------------------------------
__global__ void exp_seg_k(float* __restrict__ ap, const float* __restrict__ mx,
                          const int* __restrict__ dst, float* __restrict__ z) {
    int i = blockIdx.x * 256 + threadIdx.x;
    if (i >= NE * DD) return;
    int e = i / DD, d = i - e * DD;
    float v = __expf(ap[i] - mx[d]);
    ap[i] = v;
    unsafeAtomicAdd(&z[(size_t)dst[e] * DD + d], v);
}

__global__ void exp_only_k(float* __restrict__ ar, const float* __restrict__ mx) {
    int i = blockIdx.x * 256 + threadIdx.x;
    if (i >= NE * DD) return;
    int d = i % DD;
    ar[i] = __expf(ar[i] - mx[d]);
}

// ---------------------------------------------------------------------------
// a = ap / (z[dst] + eps);  z2[dst] += a * ex
// ---------------------------------------------------------------------------
__global__ void agg_k(const float* __restrict__ ap, const float* __restrict__ z,
                      const float* __restrict__ ex, const int* __restrict__ dst,
                      float* __restrict__ z2) {
    int i = blockIdx.x * 256 + threadIdx.x;
    if (i >= NE * DD) return;
    int e = i / DD, d = i - e * DD;
    size_t zo = (size_t)dst[e] * DD + d;
    float a = ap[i] / (z[zo] + 1e-5f);
    unsafeAtomicAdd(&z2[zo], a * ex[i]);
}

// ---------------------------------------------------------------------------
// blend = (ar/(ap+ar))*nx[src] + (ap/(ap+ar))*nx[dst];  cat = [blend, ex] bf16
// ---------------------------------------------------------------------------
__global__ void blend_cat_k(const float* __restrict__ ap, const float* __restrict__ ar,
                            const float* __restrict__ nx, const float* __restrict__ ex,
                            const int* __restrict__ src, const int* __restrict__ dst,
                            unsigned short* __restrict__ cat) {
    int i = blockIdx.x * 256 + threadIdx.x;
    if (i >= NE * DD) return;
    int e = i / DD, d = i - e * DD;
    float p = ap[i], r = ar[i];
    float inv = 1.0f / (p + r);
    float b = r * inv * nx[(size_t)src[e] * DD + d] + p * inv * nx[(size_t)dst[e] * DD + d];
    size_t o = (size_t)e * K2 + d;
    cat[o]      = f2bf(b);
    cat[o + DD] = f2bf(ex[i]);
}

// ---------------------------------------------------------------------------
// node cat = [z2, nx] bf16
// ---------------------------------------------------------------------------
__global__ void node_cat_k(const float* __restrict__ z2, const float* __restrict__ nx,
                           unsigned short* __restrict__ cat) {
    int i = blockIdx.x * 256 + threadIdx.x;
    if (i >= NN * DD) return;
    int e = i / DD, d = i - e * DD;
    size_t o = (size_t)e * K2 + d;
    cat[o]      = f2bf(z2[i]);
    cat[o + DD] = f2bf(nx[i]);
}

// ---------------------------------------------------------------------------
// WMMA GEMM: C[M, 512] = A_bf16[M, K] @ Wt_bf16[512, K]^T
// 128x64 block tile; 8 waves (4m x 2n); each wave: 32 rows x 32 cols
// -> four v_wmma_f32_16x16x32_bf16 per 32-wide K step (2 A-frags x 2 B-frags).
// A/B tiles staged LDS-direct via global_load_async_to_lds_b128 (ASYNCcnt),
// double-buffered so tile k+1 transfer overlaps tile k WMMAs.
// act != 0 applies double-LeakyReLU(0.1) on store.
// ---------------------------------------------------------------------------
__global__ __launch_bounds__(256) void wmma_gemm_k(
    const unsigned short* __restrict__ A, const unsigned short* __restrict__ Bt,
    float* __restrict__ C, int M, int K, int act)
{
    __shared__ unsigned short sA[2][128][40];  // 40*2B row stride: 16B-aligned, skewed
    __shared__ unsigned short sB[2][64][40];

    const int bm   = blockIdx.x * 128;
    const int bn   = blockIdx.y * 64;
    const int tid  = threadIdx.x;
    const int lane = tid & 31;
    const int wave = tid >> 5;
    const int wm   = wave >> 1;          // 0..3 : 32-row band
    const int wn   = wave & 1;           // 0..1 : 32-col band
    const int half = lane >> 4;          // 0/1
    const int l15  = lane & 15;

    v8f acc00 = {}, acc01 = {}, acc10 = {}, acc11 = {};

#if HAVE_ASYNC
    // per-thread async staging: A = 2 x b128 (512 segs), B = 1 x b128 (256 segs)
    auto issue_tile = [&](int buf, int k0) {
#pragma unroll
        for (int j = 0; j < 2; ++j) {
            int s   = tid + j * 256;
            int row = s >> 2;            // 0..127
            int col = (s & 3) * 8;       // bf16 offset, 16B chunk
            int gr  = bm + row;
            if (gr >= M) gr = M - 1;     // clamp: tail rows never stored
            __builtin_amdgcn_global_load_async_to_lds_b128(
                AS1P(A + (size_t)gr * K + k0 + col),
                AS3P(&sA[buf][row][col]), 0, 0);
        }
        {
            int row = tid >> 2;          // 0..63
            int col = (tid & 3) * 8;
            __builtin_amdgcn_global_load_async_to_lds_b128(
                AS1P(Bt + (size_t)(bn + row) * K + k0 + col),
                AS3P(&sB[buf][row][col]), 0, 0);
        }
    };

    issue_tile(0, 0);
    int buf = 0;
    for (int k0 = 0; k0 < K; k0 += 32) {
        if (k0 + 32 < K) {
            issue_tile(buf ^ 1, k0 + 32);  // 3 newer ops outstanding
            WAIT_ASYNC(3);                 // oldest 3 (current tile) landed
        } else {
            WAIT_ASYNC(0);
        }
        __syncthreads();
#else
    const int buf  = 0;
    const int lrowB = tid >> 2;
    const int lseg  = (tid & 3) * 8;
    for (int k0 = 0; k0 < K; k0 += 32) {
#pragma unroll
        for (int j = 0; j < 2; ++j) {
            int s   = tid + j * 256;
            int row = s >> 2;
            int col = (s & 3) * 8;
            int gr  = bm + row;
            if (gr >= M) gr = M - 1;
            *(uint4*)&sA[0][row][col] = *(const uint4*)(A + (size_t)gr * K + k0 + col);
        }
        *(uint4*)&sB[0][lrowB][lseg] =
            *(const uint4*)(Bt + (size_t)(bn + lrowB) * K + k0 + lseg);
        if (k0 + 32 < K) {
            __builtin_prefetch(A + (size_t)(bm + (tid >> 2)) * K + k0 + 32, 0, 3);
            __builtin_prefetch(Bt + (size_t)(bn + lrowB) * K + k0 + 32, 0, 3);
        }
        __syncthreads();
#endif
        // A fragment (16x32, row M = l15): lanes 0-15 hold K {0..7,16..23},
        // lanes 16-31 hold K {8..15,24..31}  -> base = half*8
        FragU fa0, fa1, fb0, fb1;
        fa0.q[0] = *(const uint4*)&sA[buf][wm * 32 + l15][half * 8];
        fa0.q[1] = *(const uint4*)&sA[buf][wm * 32 + l15][half * 8 + 16];
        fa1.q[0] = *(const uint4*)&sA[buf][wm * 32 + 16 + l15][half * 8];
        fa1.q[1] = *(const uint4*)&sA[buf][wm * 32 + 16 + l15][half * 8 + 16];
        // B fragment (32x16, col N = l15): lanes 0-15 hold K {0..7,16..23},
        // lanes 16-31 hold K {8..15,24..31} -> base = half*8 (matches A)
        fb0.q[0] = *(const uint4*)&sB[buf][wn * 32 + l15][half * 8];
        fb0.q[1] = *(const uint4*)&sB[buf][wn * 32 + l15][half * 8 + 16];
        fb1.q[0] = *(const uint4*)&sB[buf][wn * 32 + 16 + l15][half * 8];
        fb1.q[1] = *(const uint4*)&sB[buf][wn * 32 + 16 + l15][half * 8 + 16];

        acc00 = __builtin_amdgcn_wmma_f32_16x16x32_bf16(false, fa0.v, false, fb0.v,
                                                        (short)0, acc00, false, false);
        acc01 = __builtin_amdgcn_wmma_f32_16x16x32_bf16(false, fa0.v, false, fb1.v,
                                                        (short)0, acc01, false, false);
        acc10 = __builtin_amdgcn_wmma_f32_16x16x32_bf16(false, fa1.v, false, fb0.v,
                                                        (short)0, acc10, false, false);
        acc11 = __builtin_amdgcn_wmma_f32_16x16x32_bf16(false, fa1.v, false, fb1.v,
                                                        (short)0, acc11, false, false);
        __syncthreads();
#if HAVE_ASYNC
        buf ^= 1;
#endif
    }

    // C/D layout: VGPR i, lanes 0-15 -> (M=i, N=lane); lanes 16-31 -> (M=i+8)
    const int col0 = bn + wn * 32 + l15;
    const int rb0  = bm + wm * 32 + half * 8;
#pragma unroll
    for (int i = 0; i < 8; ++i) {
        int r0 = rb0 + i;
        int r1 = r0 + 16;
        if (r0 < M) {
            float v0 = acc00[i], v1 = acc01[i];
            if (act) { v0 = leaky2(v0); v1 = leaky2(v1); }
            C[(size_t)r0 * DD + col0]      = v0;
            C[(size_t)r0 * DD + col0 + 16] = v1;
        }
        if (r1 < M) {
            float v0 = acc10[i], v1 = acc11[i];
            if (act) { v0 = leaky2(v0); v1 = leaky2(v1); }
            C[(size_t)r1 * DD + col0]      = v0;
            C[(size_t)r1 * DD + col0 + 16] = v1;
        }
    }
}

// ---------------------------------------------------------------------------
// tiny output heads: out[M,N] = X[M,512] @ W[512,N]   (N = 55 or 2)
// ---------------------------------------------------------------------------
__global__ void small_gemm_k(const float* __restrict__ X, const float* __restrict__ W,
                             float* __restrict__ out, int M, int N) {
    int i = blockIdx.x * 256 + threadIdx.x;
    if (i >= M * N) return;
    int m = i / N, n = i - m * N;
    const float* xr = X + (size_t)m * DD;
    float s = 0.0f;
    for (int k = 0; k < DD; ++k) s += xr[k] * W[(size_t)k * N + n];
    out[i] = s;
}

// ---------------------------------------------------------------------------
extern "C" void kernel_launch(void* const* d_in, const int* in_sizes, int n_in,
                              void* d_out, int out_size, void* d_ws, size_t ws_size,
                              hipStream_t stream) {
    const float* node_feats = (const float*)d_in[0];
    const float* edge_feats = (const float*)d_in[1];
    const int*   src        = (const int*)d_in[2];
    const int*   dst        = (const int*)d_in[3];
    const float* Wa         = (const float*)d_in[4];   // [L, 1024, 512]
    const float* Wn         = (const float*)d_in[5];
    const float* We         = (const float*)d_in[6];
    const float* Wnc        = (const float*)d_in[7];   // [512, 55]
    const float* Wec        = (const float*)d_in[8];   // [512, 2]

    // ---- workspace layout ----
    char*  ws  = (char*)d_ws;
    size_t off = 0;
    auto alloc = [&](size_t bytes) -> void* {
        void* p = ws + off;
        off += (bytes + 255) & ~(size_t)255;
        return p;
    };
    float* nx   = (float*)alloc((size_t)NN * DD * 4);
    float* ex   = (float*)alloc((size_t)NE * DD * 4);
    float* ap   = (float*)alloc((size_t)NE * DD * 4);
    float* ar   = (float*)alloc((size_t)NE * DD * 4);
    float* z    = (float*)alloc((size_t)NN * DD * 4);
    float* z2   = (float*)alloc((size_t)NN * DD * 4);
    float* maxb = (float*)alloc(2 * DD * 4);
    unsigned short* Wat  = (unsigned short*)alloc((size_t)K2 * DD * 2);
    unsigned short* Wnt  = (unsigned short*)alloc((size_t)K2 * DD * 2);
    unsigned short* Wet  = (unsigned short*)alloc((size_t)K2 * DD * 2);
    unsigned short* catb = (unsigned short*)alloc((size_t)NE * K2 * 2);

    (void)hipMemcpyAsync(nx, node_feats, (size_t)NN * DD * 4, hipMemcpyDeviceToDevice, stream);
    (void)hipMemcpyAsync(ex, edge_feats, (size_t)NE * DD * 4, hipMemcpyDeviceToDevice, stream);

    const int ED_BLKS = (NE * DD) / 256;   // 160000
    const int ND_BLKS = (NN * DD) / 256;   // 20000
    const int TC_BLKS = (K2 * DD) / 256;   // 2048
    const dim3 GE(NE / 128, DD / 64);                 // 625 x 8
    const dim3 GN((NN + 127) / 128, DD / 64);         // 79 x 8

    for (int l = 0; l < LL; ++l) {
        const size_t wo = (size_t)l * K2 * DD;
        tcast_k<<<TC_BLKS, 256, 0, stream>>>(Wa + wo, Wat, K2, DD);
        tcast_k<<<TC_BLKS, 256, 0, stream>>>(Wn + wo, Wnt, K2, DD);
        tcast_k<<<TC_BLKS, 256, 0, stream>>>(We + wo, Wet, K2, DD);

        fill_k<<<(NN * DD + 255) / 256, 256, 0, stream>>>(z, 0.0f, NN * DD);
        fill_k<<<(NN * DD + 255) / 256, 256, 0, stream>>>(z2, 0.0f, NN * DD);
        fill_k<<<(2 * DD + 255) / 256, 256, 0, stream>>>(maxb, -__builtin_huge_valf(), 2 * DD);

        // ap = [ex, nx[dst]] @ Wa
        cat_edge_k<<<ED_BLKS, 256, 0, stream>>>(ex, nx, dst, catb);
        wmma_gemm_k<<<GE, 256, 0, stream>>>(catb, Wat, ap, NE, K2, 0);
        colmax_k<<<(NE + 255) / 256, 256, 0, stream>>>(ap, maxb, NE);

        // ar = [ex, nx[src]] @ Wa
        cat_edge_k<<<ED_BLKS, 256, 0, stream>>>(ex, nx, src, catb);
        wmma_gemm_k<<<GE, 256, 0, stream>>>(catb, Wat, ar, NE, K2, 0);
        colmax_k<<<(NE + 255) / 256, 256, 0, stream>>>(ar, maxb + DD, NE);

        // softmax pieces + aggregation
        exp_seg_k<<<ED_BLKS, 256, 0, stream>>>(ap, maxb, dst, z);
        exp_only_k<<<ED_BLKS, 256, 0, stream>>>(ar, maxb + DD);
        agg_k<<<ED_BLKS, 256, 0, stream>>>(ap, z, ex, dst, z2);

        // edge update (uses OLD nx, OLD ex)
        blend_cat_k<<<ED_BLKS, 256, 0, stream>>>(ap, ar, nx, ex, src, dst, catb);
        wmma_gemm_k<<<GE, 256, 0, stream>>>(catb, Wet, ex, NE, K2, 1);

        // node update (uses OLD nx)
        node_cat_k<<<ND_BLKS, 256, 0, stream>>>(z2, nx, catb);
        wmma_gemm_k<<<GN, 256, 0, stream>>>(catb, Wnt, nx, NN, K2, 1);
    }

    // outputs: (nx, node_conf, ex, edge_conf) flat
    float* out  = (float*)d_out;
    float* o_nx = out;
    float* o_nc = o_nx + (size_t)NN * DD;
    float* o_ex = o_nc + (size_t)NN * 55;
    float* o_ec = o_ex + (size_t)NE * DD;

    (void)hipMemcpyAsync(o_nx, nx, (size_t)NN * DD * 4, hipMemcpyDeviceToDevice, stream);
    (void)hipMemcpyAsync(o_ex, ex, (size_t)NE * DD * 4, hipMemcpyDeviceToDevice, stream);
    small_gemm_k<<<(NN * 55 + 255) / 256, 256, 0, stream>>>(nx, Wnc, o_nc, NN, 55);
    small_gemm_k<<<(NE * 2 + 255) / 256, 256, 0, stream>>>(ex, Wec, o_ec, NE, 2);
}
